// GCN_83468394430688
// MI455X (gfx1250) — matrix operand order repaired and checked by reading, not compiled, not run on hardware
//
#include <hip/hip_runtime.h>
#include <hip/hip_bf16.h>

// ---------------- types ----------------
typedef __attribute__((ext_vector_type(16))) __bf16 v16bf;
typedef __attribute__((ext_vector_type(8)))  __bf16 v8bf;
typedef __attribute__((ext_vector_type(4)))  __bf16 v4bf;
typedef __attribute__((ext_vector_type(8)))  float  v8f;
typedef __attribute__((ext_vector_type(4)))  float  v4f;

#define HID 128

__device__ __forceinline__ v16bf join8(v8bf lo, v8bf hi) {
  return __builtin_shufflevector(lo, hi, 0,1,2,3,4,5,6,7,8,9,10,11,12,13,14,15);
}

// ---------------- utility kernels ----------------
__global__ void fill_f32(float* __restrict__ p, float v, long long n) {
  long long i = ((long long)blockIdx.x * blockDim.x + threadIdx.x) * 4;
  if (i + 3 < n) {
    v4f o = {v, v, v, v};
    *(v4f*)(p + i) = o;
  } else {
    for (long long j = i; j < n; ++j) p[j] = v;
  }
}

__global__ void count_deg(const int* __restrict__ src, const int* __restrict__ dst,
                          float* __restrict__ outdeg, float* __restrict__ indeg, int E) {
  int e = blockIdx.x * blockDim.x + threadIdx.x;
  if (e < E) {
    atomicAdd(&outdeg[src[e]], 1.0f);
    atomicAdd(&indeg[dst[e]], 1.0f);
  }
}

__global__ void make_norms(const float* __restrict__ outdeg, const float* __restrict__ indeg,
                           float* __restrict__ out_norm, float* __restrict__ in_norm, int n) {
  int i = blockIdx.x * blockDim.x + threadIdx.x;
  if (i < n) {
    out_norm[i] = rsqrtf(fmaxf(outdeg[i], 1.0f));
    in_norm[i]  = rsqrtf(fmaxf(indeg[i], 1.0f));
  }
}

// f32 -> bf16, optionally scaling each row (HID cols) by rowscale[row]
__global__ void cvt_bf16_rowscale(const float* __restrict__ in,
                                  const float* __restrict__ rowscale,
                                  __bf16* __restrict__ out, long long total) {
  long long i = ((long long)blockIdx.x * blockDim.x + threadIdx.x) * 4;
  if (i >= total) return;
  v4f v = *(const v4f*)(in + i);
  float s = rowscale ? rowscale[i / HID] : 1.0f;  // 128 % 4 == 0 -> same row
  v4bf o;
  o.x = (__bf16)(v.x * s);
  o.y = (__bf16)(v.y * s);
  o.z = (__bf16)(v.z * s);
  o.w = (__bf16)(v.w * s);
  *(v4bf*)(out + i) = o;
}

// one wave (32 lanes) per edge, 4 cols per lane; agg[dst] += h[src]*out_norm[src]
__global__ void scatter_edges(const __bf16* __restrict__ h, const int* __restrict__ src,
                              const int* __restrict__ dst, const float* __restrict__ out_norm,
                              float* __restrict__ agg, int E) {
  int e = blockIdx.x * 8 + (threadIdx.x >> 5);
  int lane = threadIdx.x & 31;
  if (e >= E) return;
  int s = src[e], d = dst[e];
  float on = out_norm[s];
  v4bf hv = *(const v4bf*)(h + (long long)s * HID + lane * 4);
  float* ap = agg + (long long)d * HID + lane * 4;
  atomicAdd(ap + 0, (float)hv.x * on);
  atomicAdd(ap + 1, (float)hv.y * on);
  atomicAdd(ap + 2, (float)hv.z * on);
  atomicAdd(ap + 3, (float)hv.w * on);
}

// ---------------- WMMA GEMM: out[M x NC] = (relu?)(A[M x 128] @ W[128 x NC] + b) ----
// 128 threads = 4 waves; each wave: 16 rows x NTILES*16 cols. K = 128 (4 wmma steps).
template <int NTILES, bool RELU, bool OUTBF16>
__global__ __launch_bounds__(128) void gemm_wmma(
    const __bf16* __restrict__ A, const float* __restrict__ W,
    const float* __restrict__ bias, void* __restrict__ outp, int M, int NC) {
  constexpr int NPAD = NTILES * 16;
  __shared__ __bf16 sW[NPAD * HID];  // transposed: sW[n*128 + k]

  int tid = threadIdx.x;
  for (int idx = tid; idx < NPAD * HID; idx += 128) {
    int ncol = idx >> 7;     // output column
    int k    = idx & 127;    // K index (contiguous in LDS)
    sW[idx] = (ncol < NC) ? (__bf16)W[k * NC + ncol] : (__bf16)0.0f;
  }
  __syncthreads();

  int wave = tid >> 5;
  int lane = tid & 31;
  int hi   = lane >> 4;            // 0: lanes 0-15, 1: lanes 16-31
  int l16  = lane & 15;
  int rowBase = (blockIdx.x * 4 + wave) * 16;

  int arow = rowBase + l16;
  if (arow >= M) arow = M - 1;     // clamp loads; stores are guarded
  const __bf16* aRow = A + (long long)arow * HID;

  v8f acc[NTILES];
#pragma unroll
  for (int t = 0; t < NTILES; ++t) {
    int col = t * 16 + l16;
    float bv = (col < NC) ? bias[col] : 0.0f;
    acc[t] = (v8f){bv, bv, bv, bv, bv, bv, bv, bv};
  }

#pragma unroll
  for (int kk = 0; kk < HID; kk += 32) {
    // A fragment: 16x32 bf16, row fixed per lane, 16 contiguous K (k = 16*hi + s)
    int kbA = kk + hi * 16;
    v8bf a0 = *(const v8bf*)(aRow + kbA);
    v8bf a1 = *(const v8bf*)(aRow + kbA + 8);
    v16bf aF = join8(a0, a1);
    // B fragments from LDS (col fixed per lane, 16 contiguous K)
    int kbB = kk + hi * 16;
#pragma unroll
    for (int t = 0; t < NTILES; ++t) {
      const __bf16* wp = &sW[(t * 16 + l16) * HID + kbB];
      v16bf bF = join8(*(const v8bf*)wp, *(const v8bf*)(wp + 8));
      acc[t] = __builtin_amdgcn_wmma_f32_16x16x32_bf16(
          false, aF, false, bF, (short)0, acc[t], false, false);
    }
  }

#pragma unroll
  for (int t = 0; t < NTILES; ++t) {
    int col = t * 16 + l16;
    if (col >= NC) continue;
#pragma unroll
    for (int v = 0; v < 8; ++v) {
      int row = rowBase + hi * 8 + v;  // C layout: lanes 0-15 -> M=v, 16-31 -> M=8+v
      if (row >= M) continue;
      float val = acc[t][v];
      if (RELU) val = fmaxf(val, 0.0f);
      if (OUTBF16)
        ((__bf16*)outp)[(long long)row * NC + col] = (__bf16)val;
      else
        ((float*)outp)[(long long)row * NC + col] = val;
    }
  }
}

// ---------------- host-side orchestration ----------------
extern "C" void kernel_launch(void* const* d_in, const int* in_sizes, int n_in,
                              void* d_out, int out_size, void* d_ws, size_t ws_size,
                              hipStream_t stream) {
  const float* x  = (const float*)d_in[0];
  const int* src  = (const int*)d_in[1];
  const int* dst  = (const int*)d_in[2];
  const float* W1 = (const float*)d_in[3];
  const float* b1 = (const float*)d_in[4];
  const float* W2 = (const float*)d_in[5];
  const float* b2 = (const float*)d_in[6];
  const float* Wg = (const float*)d_in[7];
  const float* bg = (const float*)d_in[8];
  const float* Wc = (const float*)d_in[9];
  const float* bc = (const float*)d_in[10];

  const int N = in_sizes[0] / HID;      // 100000
  const int E = in_sizes[1];            // 1600000
  const int NCLASS = in_sizes[10];      // 40
  const int NLAYER = in_sizes[8] / HID; // 3

  // workspace carve-up (256B aligned)
  char* ws = (char*)d_ws;
  size_t off = 0;
  auto carve = [&](size_t bytes) {
    char* p = ws + off;
    off = (off + bytes + 255) & ~(size_t)255;
    return p;
  };
  float*  outdeg   = (float*)carve((size_t)N * 4);
  float*  indeg    = (float*)carve((size_t)N * 4);
  float*  out_norm = (float*)carve((size_t)N * 4);
  float*  in_norm  = (float*)carve((size_t)N * 4);
  __bf16* bufA     = (__bf16*)carve((size_t)N * HID * 2);
  __bf16* bufB     = (__bf16*)carve((size_t)N * HID * 2);
  float*  agg      = (float*)carve((size_t)N * HID * 4);
  (void)ws_size; (void)n_in; (void)out_size;

  const long long NH = (long long)N * HID;
  const int fillN_blk  = (int)((N + 4 * 256 - 1) / (4 * 256));
  const int fillNH_blk = (int)((NH + 4 * 256 - 1) / (4 * 256));
  const int cvt_blk    = fillNH_blk;
  const int gemm_blk   = (N + 63) / 64;

  // degrees -> norms
  fill_f32<<<fillN_blk, 256, 0, stream>>>(outdeg, 0.0f, N);
  fill_f32<<<fillN_blk, 256, 0, stream>>>(indeg, 0.0f, N);
  count_deg<<<(E + 255) / 256, 256, 0, stream>>>(src, dst, outdeg, indeg, E);
  make_norms<<<(N + 255) / 256, 256, 0, stream>>>(outdeg, indeg, out_norm, in_norm, N);

  // h = (x@W1+b1)@W2+b2  (bf16 activations, f32 accumulate)
  cvt_bf16_rowscale<<<cvt_blk, 256, 0, stream>>>(x, nullptr, bufA, NH);
  gemm_wmma<8, false, true><<<gemm_blk, 128, 0, stream>>>(bufA, W1, b1, bufB, N, HID);
  gemm_wmma<8, false, true><<<gemm_blk, 128, 0, stream>>>(bufB, W2, b2, bufA, N, HID);

  // 3x GraphConv + ReLU; h lives in bufA at loop top
  for (int l = 0; l < NLAYER; ++l) {
    fill_f32<<<fillNH_blk, 256, 0, stream>>>(agg, 0.0f, NH);
    scatter_edges<<<(E + 7) / 8, 256, 0, stream>>>(bufA, src, dst, out_norm, agg, E);
    cvt_bf16_rowscale<<<cvt_blk, 256, 0, stream>>>(agg, in_norm, bufB, NH);
    gemm_wmma<8, true, true><<<gemm_blk, 128, 0, stream>>>(
        bufB, Wg + (size_t)l * HID * HID, bg + (size_t)l * HID, bufA, N, HID);
  }

  // classifier: f32 output, 40 cols (3 tiles, masked)
  gemm_wmma<3, false, false><<<gemm_blk, 128, 0, stream>>>(bufA, Wc, bc, d_out, N, NCLASS);
}
